// GraphEncoder_51866025066751
// MI455X (gfx1250) — compile-verified, matrix operand order checked by reading, not run on hardware
//
#include <hip/hip_runtime.h>
#include <hip/hip_bf16.h>

#define N_NODES 50000
#define N_EDGES 800000
#define HID 64
#define N_GRAPHS 64

typedef __attribute__((ext_vector_type(2)))  float v2f;
typedef __attribute__((ext_vector_type(8)))  float v8f;
typedef __attribute__((ext_vector_type(16))) _Float16 v16h;

#ifndef __has_builtin
#define __has_builtin(x) 0
#endif

#if __has_builtin(__builtin_amdgcn_wmma_f32_16x16x4_f32)
#define USE_F32_WMMA 1
#else
#define USE_F32_WMMA 0
#endif

// -------------------- degree / dinv --------------------
__global__ void k_init_one(float* __restrict__ p, int n) {
    int i = blockIdx.x * blockDim.x + threadIdx.x;
    if (i < n) p[i] = 1.0f;
}
__global__ void k_deg_edges(const int* __restrict__ dst, float* __restrict__ deg, int E) {
    int e = blockIdx.x * blockDim.x + threadIdx.x;
    if (e < E) atomicAdd(&deg[dst[e]], 1.0f);
}
__global__ void k_dinv(float* __restrict__ deg, int n) {
    int i = blockIdx.x * blockDim.x + threadIdx.x;
    if (i < n) deg[i] = rsqrtf(deg[i]);
}

// ---------------- WMMA GEMM, one wave per 16-row x 64-col block -----------------
// H = A @ W           (A is [rows,K] f32, optionally row-gathered via gatherIdx)
// if Xout != nullptr:  X = H * dinv[row]^2 + bias   (GCN self-loop + bias, fused)
// else:                H += bias                    (MLP mode)
// rows must be a multiple of 16 (50000 = 3125*16, 64 = 4*16).
__global__ void k_gemm_fused(const float* __restrict__ A, const float* __restrict__ W,
                             const float* __restrict__ bias,
                             float* __restrict__ Hout, float* __restrict__ Xout,
                             const float* __restrict__ dinv,
                             const int* __restrict__ gatherIdx,
                             int K, int nTiles) {
    int tile = blockIdx.x * (blockDim.x >> 5) + (threadIdx.x >> 5);
    if (tile >= nTiles) return;           // uniform per wave -> EXEC stays all-1s
    int r0   = tile * 16;
    int lane = threadIdx.x & 31;
    int half = lane >> 4;                 // 0: K+0/1 slice, 1: K+2/3 slice
    int l15  = lane & 15;

    // A-fragment source row (optional gather for the embedding layer)
    int arowIdx = r0 + l15;
    if (gatherIdx) arowIdx = gatherIdx[arowIdx];
    const float* arowBase = A + (size_t)arowIdx * K;

    v8f acc0 = {}, acc1 = {}, acc2 = {}, acc3 = {};
#if USE_F32_WMMA
    for (int k0 = 0; k0 < K; k0 += 4) {
        int ka = k0 + half * 2;
        v2f a;  a.x  = arowBase[ka];        a.y  = arowBase[ka + 1];
        const float* wp = W + (size_t)ka * HID + l15;
        v2f b0; b0.x = wp[0];               b0.y = wp[HID];
        v2f b1; b1.x = wp[16];              b1.y = wp[16 + HID];
        v2f b2; b2.x = wp[32];              b2.y = wp[32 + HID];
        v2f b3; b3.x = wp[48];              b3.y = wp[48 + HID];
        acc0 = __builtin_amdgcn_wmma_f32_16x16x4_f32(false, a, false, b0, (short)0, acc0, false, false);
        acc1 = __builtin_amdgcn_wmma_f32_16x16x4_f32(false, a, false, b1, (short)0, acc1, false, false);
        acc2 = __builtin_amdgcn_wmma_f32_16x16x4_f32(false, a, false, b2, (short)0, acc2, false, false);
        acc3 = __builtin_amdgcn_wmma_f32_16x16x4_f32(false, a, false, b3, (short)0, acc3, false, false);
    }
#else
    // f16 fallback (codegen-confirmed builtin); compile-only safety net
    for (int k0 = 0; k0 < K; k0 += 32) {
        v16h a = {};
        for (int v = 0; v < 8; ++v) {
            int kk = k0 + ((v < 4) ? (2 * v) : (16 + 2 * (v - 4))) + half * 8;
            if (kk < K)     a[2 * v]     = (_Float16)arowBase[kk];
            if (kk + 1 < K) a[2 * v + 1] = (_Float16)arowBase[kk + 1];
        }
        for (int c = 0; c < 4; ++c) {
            v16h b = {};
            for (int v = 0; v < 8; ++v) {
                int kk = k0 + ((v < 4) ? (2 * v) : (16 + 2 * (v - 4))) + half * 8;
                if (kk < K)     b[2 * v]     = (_Float16)W[(size_t)kk * HID + c * 16 + l15];
                if (kk + 1 < K) b[2 * v + 1] = (_Float16)W[(size_t)(kk + 1) * HID + c * 16 + l15];
            }
            v8f* accp = (c == 0) ? &acc0 : (c == 1) ? &acc1 : (c == 2) ? &acc2 : &acc3;
            *accp = __builtin_amdgcn_wmma_f32_16x16x32_f16(false, a, false, b, (short)0, *accp, false, false);
        }
    }
#endif
    // epilogue: C/D layout -> VGPR v holds row r0+v (lanes 0-15) / r0+v+8 (lanes 16-31)
    float bias4[4];
    #pragma unroll
    for (int c = 0; c < 4; ++c) bias4[c] = bias[c * 16 + l15];

    #pragma unroll
    for (int v = 0; v < 8; ++v) {
        int row = r0 + v + half * 8;
        size_t base = (size_t)row * HID + l15;
        float vals[4] = { acc0[v], acc1[v], acc2[v], acc3[v] };
        if (Xout) {
            float di = dinv[row];
            float dd = di * di;
            #pragma unroll
            for (int c = 0; c < 4; ++c) {
                float hv = vals[c];
                Hout[base + c * 16] = hv;
                Xout[base + c * 16] = hv * dd + bias4[c];
            }
        } else {
            #pragma unroll
            for (int c = 0; c < 4; ++c)
                Hout[base + c * 16] = vals[c] + bias4[c];
        }
    }
}

// ------ edge scatter: X[dst] += H[src] * dinv[s]*dinv[d], float4 per thread ------
__global__ void k_edge_scatter(const int* __restrict__ src, const int* __restrict__ dst,
                               const float* __restrict__ H, const float* __restrict__ dinv,
                               float* __restrict__ X, int E) {
    int idx = blockIdx.x * blockDim.x + threadIdx.x;
    if (idx >= E * 16) return;
    int e = idx >> 4;                 // 16 consecutive threads per edge
    int j = (idx & 15) * 4;
    int s = src[e], d = dst[e];
    float coef = dinv[s] * dinv[d];
    const float4 hv = *(const float4*)(H + (size_t)s * HID + j);
    float* xp = X + (size_t)d * HID + j;
    atomicAdd(xp + 0, hv.x * coef);
    atomicAdd(xp + 1, hv.y * coef);
    atomicAdd(xp + 2, hv.z * coef);
    atomicAdd(xp + 3, hv.w * coef);
}

// -------------------- fused ReLU + LayerNorm (one wave per node) --------------------
__global__ void k_relu_ln(float* __restrict__ x, const float* __restrict__ g,
                          const float* __restrict__ b, int n) {
    int node = blockIdx.x * (blockDim.x >> 5) + (threadIdx.x >> 5);
    if (node >= n) return;
    int lane = threadIdx.x & 31;
    float* row = x + (size_t)node * HID;
    float v0 = fmaxf(row[lane], 0.0f);
    float v1 = fmaxf(row[lane + 32], 0.0f);
    float s = v0 + v1;
    for (int m = 16; m; m >>= 1) s += __shfl_xor(s, m, 32);
    float mu = s * (1.0f / 64.0f);
    float d0 = v0 - mu, d1 = v1 - mu;
    float vs = d0 * d0 + d1 * d1;
    for (int m = 16; m; m >>= 1) vs += __shfl_xor(vs, m, 32);
    float rstd = rsqrtf(vs * (1.0f / 64.0f) + 1e-5f);
    row[lane]      = d0 * rstd * g[lane]      + b[lane];
    row[lane + 32] = d1 * rstd * g[lane + 32] + b[lane + 32];
}

// -------------------- pooling --------------------
__global__ void k_zero(float* __restrict__ p, int n) {
    int i = blockIdx.x * blockDim.x + threadIdx.x;
    if (i < n) p[i] = 0.0f;
}
__global__ void k_pool_scatter(const float* __restrict__ x, const int* __restrict__ batch,
                               float* __restrict__ psum, float* __restrict__ cnt, int n) {
    int idx = blockIdx.x * blockDim.x + threadIdx.x;
    if (idx >= n * 16) return;
    int i = idx >> 4;
    int j = (idx & 15) * 4;
    int gidx = batch[i];
    const float4 xv = *(const float4*)(x + (size_t)i * HID + j);
    float* pp = psum + (size_t)gidx * HID + j;
    atomicAdd(pp + 0, fmaxf(xv.x, 0.0f));   // ReLU fused
    atomicAdd(pp + 1, fmaxf(xv.y, 0.0f));
    atomicAdd(pp + 2, fmaxf(xv.z, 0.0f));
    atomicAdd(pp + 3, fmaxf(xv.w, 0.0f));
    if ((idx & 15) == 0) atomicAdd(&cnt[gidx], 1.0f);
}
__global__ void k_pool_div(const float* __restrict__ psum, const float* __restrict__ cnt,
                           float* __restrict__ pooled) {
    int idx = blockIdx.x * blockDim.x + threadIdx.x;
    if (idx >= N_GRAPHS * HID) return;
    pooled[idx] = psum[idx] / fmaxf(cnt[idx >> 6], 1.0f);
}

extern "C" void kernel_launch(void* const* d_in, const int* in_sizes, int n_in,
                              void* d_out, int out_size, void* d_ws, size_t ws_size,
                              hipStream_t stream) {
    const int*   node  = (const int*)d_in[0];
    const int*   src   = (const int*)d_in[1];
    const int*   dst   = (const int*)d_in[2];
    const int*   batch = (const int*)d_in[3];
    const float* emb   = (const float*)d_in[4];
    const float* W1    = (const float*)d_in[5];
    const float* b1    = (const float*)d_in[6];
    const float* W2    = (const float*)d_in[7];
    const float* b2    = (const float*)d_in[8];
    const float* W3    = (const float*)d_in[9];
    const float* b3    = (const float*)d_in[10];
    const float* ln1g  = (const float*)d_in[11];
    const float* ln1b  = (const float*)d_in[12];
    const float* ln2g  = (const float*)d_in[13];
    const float* ln2b  = (const float*)d_in[14];
    const float* pW1   = (const float*)d_in[15];
    const float* pb1   = (const float*)d_in[16];
    const float* pW2   = (const float*)d_in[17];
    const float* pb2   = (const float*)d_in[18];
    float* out = (float*)d_out;

    const int N = N_NODES, E = N_EDGES;
    float* ws = (float*)d_ws;
    size_t off = 0;
    float* dinv   = ws + off; off += 50048;           // N rounded up
    float* X      = ws + off; off += (size_t)N * HID; // layer activations
    float* H      = ws + off; off += (size_t)N * HID; // pre-aggregation h = x@W
    float* psum   = ws + off; off += N_GRAPHS * HID;
    float* cnt    = ws + off; off += 64;
    float* pooled = ws + off; off += N_GRAPHS * HID;
    float* h1     = ws + off; off += N_GRAPHS * HID;

    const dim3 B(256);
    const int gN  = (N + 255) / 256;
    const int gE  = (E + 255) / 256;
    const int gE4 = (E * 16 + 255) / 256;
    const int gP4 = (N * 16 + 255) / 256;
    const int gLN = (N + 7) / 8;                    // 8 waves/block, 1 wave/node
    const int nodeTiles = N / 16;                   // 3125 waves, 16x64 block each
    const int gGemmN = (nodeTiles + 7) / 8;
    const int mlpTiles = N_GRAPHS / 16;             // 4 waves
    const int gGemmM = 1;

    // degree -> dinv (shared by all 3 layers)
    k_init_one<<<gN, B, 0, stream>>>(dinv, N);
    k_deg_edges<<<gE, B, 0, stream>>>(dst, dinv, E);
    k_dinv<<<gN, B, 0, stream>>>(dinv, N);

    // ----- layer 1 (K=4, A row-gathered straight from the embedding table) -----
    k_gemm_fused<<<gGemmN, B, 0, stream>>>(emb, W1, b1, H, X, dinv, node, 4, nodeTiles);
    k_edge_scatter<<<gE4, B, 0, stream>>>(src, dst, H, dinv, X, E);
    k_relu_ln<<<gLN, B, 0, stream>>>(X, ln1g, ln1b, N);

    // ----- layer 2 (K=64) -----
    k_gemm_fused<<<gGemmN, B, 0, stream>>>(X, W2, b2, H, X, dinv, nullptr, HID, nodeTiles);
    k_edge_scatter<<<gE4, B, 0, stream>>>(src, dst, H, dinv, X, E);
    k_relu_ln<<<gLN, B, 0, stream>>>(X, ln2g, ln2b, N);

    // ----- layer 3 (K=64, no LN; ReLU fused into pooling) -----
    k_gemm_fused<<<gGemmN, B, 0, stream>>>(X, W3, b3, H, X, dinv, nullptr, HID, nodeTiles);
    k_edge_scatter<<<gE4, B, 0, stream>>>(src, dst, H, dinv, X, E);

    // ----- global mean pool -----
    k_zero<<<(N_GRAPHS * HID + 64 + 255) / 256, B, 0, stream>>>(psum, N_GRAPHS * HID + 64);
    k_pool_scatter<<<gP4, B, 0, stream>>>(X, batch, psum, cnt, N);
    k_pool_div<<<(N_GRAPHS * HID + 255) / 256, B, 0, stream>>>(psum, cnt, pooled);

    // ----- post-MP MLP (WMMA, rows=64) -----
    k_gemm_fused<<<gGemmM, B, 0, stream>>>(pooled, pW1, pb1, h1, nullptr, nullptr, nullptr, HID, mlpTiles);
    k_gemm_fused<<<gGemmM, B, 0, stream>>>(h1, pW2, pb2, out, nullptr, nullptr, nullptr, HID, mlpTiles);
}
